// BoundaryLoss_6519760355553
// MI455X (gfx1250) — compile-verified, matrix operand order checked by reading, not run on hardware
//
#include <hip/hip_runtime.h>
#include <hip/hip_bf16.h>
#include <float.h>
#include <stdint.h>

// Problem constants (fixed by reference setup_inputs)
#define B_  8
#define C_  4
#define H_  384
#define W_  384
#define HW_ (H_*W_)
#define N_  (B_*C_*H_*W_)        // 4,718,592 elements
#define P_  (B_*H_*W_)           // 1,179,648 pixels
#define NTILES_ (P_/16)          // 73,728 WMMA tiles (16 pixels x 4 channels)
#define BIGF 1.0e8f

#define TW 16                    // columns per workgroup in column pass
#define RED_BLOCKS 144
#define RED_THREADS 256
#define NWAVES_ (RED_BLOCKS*RED_THREADS/32)   // 1152 waves; 73728/1152 = 64 tiles each

typedef __attribute__((ext_vector_type(2)))  float    v2f;
typedef __attribute__((ext_vector_type(8)))  float    v8f;
typedef __attribute__((ext_vector_type(16))) _Float16 v16h;

// ---------------------------------------------------------------- feature probes
#if defined(__gfx1250__) && defined(__has_builtin)
# if __has_builtin(__builtin_amdgcn_wmma_f32_16x16x4_f32)
#  define WMMA_PATH 1            // native f32 16x16x4
# elif __has_builtin(__builtin_amdgcn_wmma_f32_16x16x32_f16)
#  define WMMA_PATH 2            // f16 inputs, f32 accumulate
# else
#  define WMMA_PATH 0
# endif
# if __has_builtin(__builtin_amdgcn_global_load_async_to_lds_b128)
#  define HAVE_ASYNC_LDS 1
# else
#  define HAVE_ASYNC_LDS 0
# endif
#else
# define WMMA_PATH 0
# define HAVE_ASYNC_LDS 0
#endif

#if HAVE_ASYNC_LDS
// Builtin signature (from hipcc diagnostic): param0 = int __vector_size(16)__ *
// in addrspace(1) (global source), param1 = same vector type in addrspace(3)
// (LDS destination), then imm offset, imm cpol.
typedef int v4i __attribute__((vector_size(16)));
typedef v4i __attribute__((address_space(1)))* glb_v4p;
typedef v4i __attribute__((address_space(3)))* lds_v4p;
__device__ __forceinline__ void async_copy16B(const float* g, float* l) {
    // CK-style addrspace casts via uintptr_t (generic->AS3 truncation keeps
    // the LDS offset per the ISA flat-LDS aperture rule).
    __builtin_amdgcn_global_load_async_to_lds_b128(
        reinterpret_cast<glb_v4p>(reinterpret_cast<uintptr_t>(g)),
        reinterpret_cast<lds_v4p>(reinterpret_cast<uintptr_t>(l)), 0, 0);
}
__device__ __forceinline__ void async_wait0() {
# if __has_builtin(__builtin_amdgcn_s_wait_asynccnt)
    __builtin_amdgcn_s_wait_asynccnt(0);
# else
    asm volatile("s_wait_asynccnt 0x0" ::: "memory");
# endif
}
#endif

// ---------------------------------------------------------------- k_zero
__global__ void k_zero(int* __restrict__ cnt) {
    if (threadIdx.x < B_*C_) cnt[threadIdx.x] = 0;
}

// ---------------------------------------------------------------- k_count
// Per-(b,c) mask population count. Each 256-thread block covers 256 contiguous
// pixels of a single image (HW_ % 256 == 0), histograms in LDS, 4 global atomics.
__global__ __launch_bounds__(256) void k_count(const int* __restrict__ tgt,
                                               int* __restrict__ cnt) {
    __shared__ int h4[C_];
    if (threadIdx.x < C_) h4[threadIdx.x] = 0;
    __syncthreads();
    const int i = blockIdx.x * 256 + threadIdx.x;
    const int b = i / HW_;                 // uniform within block
    atomicAdd(&h4[tgt[i]], 1);
    __syncthreads();
    if (threadIdx.x < C_) atomicAdd(&cnt[b*C_ + threadIdx.x], h4[threadIdx.x]);
}

// ---------------------------------------------------------------- k_rowpass
// d1[h,x] = min_y f[h,y] + (x-y)^2, exact f32 replication of the reference.
// One workgroup per (mask, row); both pos and neg fields in one pass.
__global__ __launch_bounds__(W_) void k_rowpass(const int* __restrict__ tgt,
                                                float* __restrict__ d1p,
                                                float* __restrict__ d1n) {
    const int h  = blockIdx.x;
    const int bc = blockIdx.y;
    const int b  = bc >> 2;
    const int c  = bc & 3;
    const int x  = threadIdx.x;
    __shared__ float fp[W_];
    __shared__ float fn[W_];
    const int t = tgt[(b*H_ + h)*W_ + x];
    const bool m = (t == c);
    fp[x] = m ? 0.0f : BIGF;
    fn[x] = m ? BIGF : 0.0f;
    __syncthreads();
    float bp = FLT_MAX, bn = FLT_MAX;
    for (int y = 0; y < W_; ++y) {
        const float d  = (float)(x - y);
        const float sq = d * d;
        bp = fminf(bp, fp[y] + sq);
        bn = fminf(bn, fn[y] + sq);
    }
    const int o = (bc*H_ + h)*W_ + x;
    d1p[o] = bp;
    d1n[o] = bn;
}

// ---------------------------------------------------------------- k_colpass
// d2[h,x] = min_y d1[y,x] + (h-y)^2 ; then phi = sqrt(pos)-sqrt(neg)+1 with
// degenerate-mask rules. Writes phi in place over d1p (tile is private to WG).
// Tile staging goes through GLOBAL_LOAD_ASYNC_TO_LDS_B128 when available.
__global__ __launch_bounds__(H_) void k_colpass(float* __restrict__ d1p,
                                                const float* __restrict__ d1n,
                                                const int* __restrict__ cnt) {
    const int x0  = blockIdx.x * TW;
    const int bc  = blockIdx.y;
    const int tid = threadIdx.x;          // row index h
    __shared__ __align__(16) float sp[H_*TW];   // 24 KB
    __shared__ __align__(16) float sn[H_*TW];   // 24 KB
    float*       basep = d1p + (size_t)bc * HW_;
    const float* basen = d1n + (size_t)bc * HW_;
#if HAVE_ASYNC_LDS
    #pragma unroll
    for (int j = 0; j < TW/4; ++j) {      // 4 floats (16B) per async op
        async_copy16B(basep + tid*W_ + x0 + j*4, sp + tid*TW + j*4);
        async_copy16B(basen + tid*W_ + x0 + j*4, sn + tid*TW + j*4);
    }
    async_wait0();
#else
    #pragma unroll
    for (int j = 0; j < TW; ++j) {
        sp[tid*TW + j] = basep[tid*W_ + x0 + j];
        sn[tid*TW + j] = basen[tid*W_ + x0 + j];
    }
#endif
    __syncthreads();
    float rp[TW], rn[TW];
    #pragma unroll
    for (int j = 0; j < TW; ++j) { rp[j] = FLT_MAX; rn[j] = FLT_MAX; }
    const int h = tid;
    for (int y = 0; y < H_; ++y) {
        const float d  = (float)(h - y);
        const float sq = d * d;
        #pragma unroll
        for (int j = 0; j < TW; ++j) {
            rp[j] = fminf(rp[j], sp[y*TW + j] + sq);   // LDS broadcast reads
            rn[j] = fminf(rn[j], sn[y*TW + j] + sq);
        }
    }
    const int s = cnt[bc];
    #pragma unroll
    for (int j = 0; j < TW; ++j) {
        const float pos = sqrtf(rp[j]);
        const float neg = sqrtf(rn[j]);
        const float phi = (s == 0) ? pos : ((s == HW_) ? -neg : (pos - neg + 1.0f));
        basep[h*W_ + x0 + j] = phi;
    }
}

// ---------------------------------------------------------------- WMMA helper
// Ones-B reduction trick: D[m][n] = sum_k A[m][k] * 1, so after accumulation
// sum(all elements of C) == 16 * (sum of every A element ever fed). This is
// invariant to the exact A/B/C VGPR layouts -> robust compile-only usage.
__device__ __forceinline__ void wmma_accum(float pa, float pb, v8f& acc) {
#if WMMA_PATH == 1
    v2f a = {pa, pb};
    const v2f ones = {1.0f, 1.0f};
    acc = __builtin_amdgcn_wmma_f32_16x16x4_f32(false, a, false, ones,
                                                (short)0, acc, false, false);
#elif WMMA_PATH == 2
    v16h a = {};
    a[0] = (_Float16)pa;
    a[1] = (_Float16)pb;
    v16h ones;
    #pragma unroll
    for (int i = 0; i < 16; ++i) ones[i] = (_Float16)1.0f;
    acc = __builtin_amdgcn_wmma_f32_16x16x32_f16(false, a, false, ones,
                                                 (short)0, acc, false, false);
#else
    acc[0] += pa + pb;           // host-pass / fallback parse path
#endif
}

// ---------------------------------------------------------------- k_reduce
// loss partials: each 16-pixel tile -> one WMMA. Lane L (L<16) handles pixel
// slot L channels {0,1}; lane L+16 handles the same pixel, channels {2,3}.
// Tile count is wave-uniform -> EXEC all-ones at every WMMA.
__global__ __launch_bounds__(RED_THREADS) void k_reduce(const float* __restrict__ logits,
                                                        const float* __restrict__ phi,
                                                        float* __restrict__ part) {
    const int lane   = threadIdx.x & 31;
    const int wave   = (blockIdx.x * RED_THREADS + threadIdx.x) >> 5;
    const int nwaves = NWAVES_;
    const int sub    = lane & 15;
    const int khi    = lane >> 4;                 // 0 -> channels 0,1 ; 1 -> 2,3
    v8f acc = {};
    for (int t = wave; t < NTILES_; t += nwaves) {
        const int p  = t * 16 + sub;
        const int b  = p / HW_;
        const int hw = p - b * HW_;
        const float* lg = logits + (size_t)(b * C_) * HW_ + hw;
        const float l0 = lg[0];
        const float l1 = lg[HW_];
        const float l2 = lg[2*HW_];
        const float l3 = lg[3*HW_];
        const float mx = fmaxf(fmaxf(l0, l1), fmaxf(l2, l3));
        const float e0 = __expf(l0 - mx);
        const float e1 = __expf(l1 - mx);
        const float e2 = __expf(l2 - mx);
        const float e3 = __expf(l3 - mx);
        const float inv = 1.0f / (e0 + e1 + e2 + e3);
        const float* ph = phi + (size_t)(b * C_ + 2*khi) * HW_ + hw;
        const float pa = (khi ? e2 : e0) * inv * ph[0];
        const float pb = (khi ? e3 : e1) * inv * ph[HW_];
        wmma_accum(pa, pb, acc);
        if (t + nwaves < NTILES_) {               // wave-uniform branch
            const int pn  = (t + nwaves) * 16 + sub;
            const int bn  = pn / HW_;
            const int hwn = pn - bn * HW_;
            __builtin_prefetch(logits + (size_t)(bn * C_) * HW_ + hwn, 0, 0);
        }
    }
    float s = acc[0]+acc[1]+acc[2]+acc[3]+acc[4]+acc[5]+acc[6]+acc[7];
    #pragma unroll
    for (int off = 16; off > 0; off >>= 1) s += __shfl_xor(s, off, 32);
    if (lane == 0) part[wave] = s;
}

// ---------------------------------------------------------------- k_final
// Deterministic fixed-order tree sum of per-wave partials -> scalar loss.
__global__ __launch_bounds__(256) void k_final(const float* __restrict__ part,
                                               float* __restrict__ out) {
    __shared__ float sm[256];
    float s = 0.0f;
    for (int i = threadIdx.x; i < NWAVES_; i += 256) s += part[i];
    sm[threadIdx.x] = s;
    __syncthreads();
    for (int off = 128; off > 0; off >>= 1) {
        if (threadIdx.x < off) sm[threadIdx.x] += sm[threadIdx.x + off];
        __syncthreads();
    }
    if (threadIdx.x == 0) out[0] = sm[0] * (1.0f/16.0f) * (1.0f/(float)N_);
}

// ---------------------------------------------------------------- launch
extern "C" void kernel_launch(void* const* d_in, const int* in_sizes, int n_in,
                              void* d_out, int out_size, void* d_ws, size_t ws_size,
                              hipStream_t stream) {
    const float* logits  = (const float*)d_in[0];   // (8,4,384,384) f32
    const int*   targets = (const int*)d_in[1];     // (8,384,384) i32
    float* out = (float*)d_out;                     // scalar f32

    // Workspace layout: [d1_pos | d1_neg | cnt(32 i32) | partials(1152 f32)]
    float* d1p  = (float*)d_ws;
    float* d1n  = d1p + (size_t)N_;
    int*   cnt  = (int*)(d1n + (size_t)N_);
    float* part = (float*)(cnt + 64);               // 64 ints keeps 256B align

    k_zero<<<1, 64, 0, stream>>>(cnt);
    k_count<<<P_/256, 256, 0, stream>>>(targets, cnt);

    dim3 g2(H_, B_*C_);
    k_rowpass<<<g2, W_, 0, stream>>>(targets, d1p, d1n);

    dim3 g3(W_/TW, B_*C_);
    k_colpass<<<g3, H_, 0, stream>>>(d1p, d1n, cnt);

    k_reduce<<<RED_BLOCKS, RED_THREADS, 0, stream>>>(logits, d1p, part);
    k_final<<<1, 256, 0, stream>>>(part, out);
}
